// ModernBertAttention_50843822850523
// MI455X (gfx1250) — compile-verified, matrix operand order checked by reading, not run on hardware
//
#include <hip/hip_runtime.h>
#include <hip/hip_bf16.h>

// Problem constants (match reference setup_inputs)
#define BB 4
#define SS 2048
#define HH 768
#define NH 12
#define HD 64

typedef __attribute__((ext_vector_type(16))) __bf16 v16bf;
typedef __attribute__((ext_vector_type(8)))  float  v8f;
typedef __attribute__((ext_vector_type(4)))  unsigned int v4u;
typedef __attribute__((ext_vector_type(8)))  int v8i;
typedef __attribute__((ext_vector_type(4)))  int v4i;

union BfFrag { v16bf v; unsigned short u[16]; };

__device__ inline unsigned short f2bf(float f) {
  unsigned u = __float_as_uint(f);
  u += 0x7FFFu + ((u >> 16) & 1u);     // round-to-nearest-even
  return (unsigned short)(u >> 16);
}
__device__ inline float bf2f(unsigned short h) {
  return __uint_as_float(((unsigned)h) << 16);
}

// K-index for element e (0..15) of a 16-bit A/B WMMA fragment (ISA 7.12.2):
// VGPR v holds K = {base, base+1}, base = (v<4?0:16) + half*8 + (v&3)*2
__device__ inline int frag_k(int e, int half) {
  int v = e >> 1;
  return ((v < 4) ? 0 : 16) + half * 8 + ((v & 3) << 1) + (e & 1);
}

__device__ inline v8f wmma_bf16(const BfFrag& a, const BfFrag& b, v8f c) {
  return __builtin_amdgcn_wmma_f32_16x16x32_bf16(
      /*neg_a=*/false, a.v, /*neg_b=*/false, b.v,
      /*c_mod=*/(short)0, c, /*reuse_a=*/false, /*reuse_b=*/false);
}

#define HAVE_TDM __has_builtin(__builtin_amdgcn_tensor_load_to_lds)

#if HAVE_TDM
// Issue one TDM descriptor: copy a (tile_h x tile_w) bf16 tile (row stride
// stride_elems in global) into LDS at lds_off, inserting pad bytes per row so
// the LDS pitch stays bank-conflict-free. One issue per wave (EXEC ignored).
// D# layout per ISA cdna5 8.3/8.4 (group0 128b, group1 256b).
// This toolchain exposes the 6-arg builtin: (v4u, v8i, v4i, v4i, v8i, cpol).
__device__ inline void tdm_load_tile_bf16(const unsigned short* gptr,
                                          unsigned lds_off,
                                          int tile_w, int tile_h,
                                          int stride_elems,
                                          int pad_interval_code,
                                          int pad_amount_code) {
  unsigned long long ga = (unsigned long long)(size_t)gptr;
  v4u g0 = {
    1u,                                             // count=1, user descriptor
    lds_off,                                        // lds_addr [63:32]
    (unsigned)(ga & 0xFFFFFFFFu),                   // global_addr lo
    (unsigned)((ga >> 32) & 0x01FFFFFFu) | (2u << 30) // global_addr hi | type=2
  };
  v8i g1 = {
    (int)((1u << 16)                                // data_size = 2 bytes
        | (1u << 20)                                // pad_enable
        | ((unsigned)pad_interval_code << 22)       // pad interval
        | ((unsigned)pad_amount_code << 25)),       // pad amount (DWORDs-1)
    (int)(((unsigned)stride_elems & 0xFFFFu) << 16),// tensor_dim0 lo16 @ [63:48]
    (int)((((unsigned)stride_elems >> 16) & 0xFFFFu)// tensor_dim0 hi16 @ [79:64]
        | (0u << 16)),                              // tensor_dim1 lo16 = 0
    (int)(16u | ((unsigned)tile_w << 16)),          // tensor_dim1 hi (=1M rows), tile_dim0
    (int)((unsigned)tile_h),                        // tile_dim1, tile_dim2=0
    stride_elems,                                   // tensor_dim0_stride lo32
    0,                                              // stride hi, dim1_stride lo
    0
  };
  v4i z4 = {0, 0, 0, 0};
  v8i z8 = {0, 0, 0, 0, 0, 0, 0, 0};
  __builtin_amdgcn_tensor_load_to_lds(g0, g1, z4, z4, z8, /*cpol=*/0);
}
#endif

// ---------------------------------------------------------------------------
// fp32 -> bf16 conversion
// ---------------------------------------------------------------------------
__global__ void f32_to_bf16_kernel(const float* __restrict__ src,
                                   unsigned short* __restrict__ dst, int n) {
  int i = blockIdx.x * blockDim.x + threadIdx.x;
  if (i < n) dst[i] = f2bf(src[i]);
}

// ---------------------------------------------------------------------------
// Generic bf16 GEMM: C[m][n] = sum_k A[m][k] * B[n][k]   (both row-major, K%32==0)
// Block tile 64x64, 8 waves (4 M x 2 N), each wave -> 16x32 of C via WMMA.
// A/B K-slices staged into LDS by the Tensor Data Mover (wave 0 issues, waits
// on TENSORcnt, block barrier publishes).
// ---------------------------------------------------------------------------
template <bool OUT_BF16>
__global__ __launch_bounds__(256)
void gemm_bf16_wmma(const unsigned short* __restrict__ A,
                    const unsigned short* __restrict__ Bm,
                    float* __restrict__ Cf, unsigned short* __restrict__ Cb,
                    int M, int N, int K) {
  __shared__ unsigned short As[64][40];   // 64 rows x 32 K (+8 pad = 16B)
  __shared__ unsigned short Bs[64][40];
  const int tid  = threadIdx.x;
  const int lane = tid & 31, wid = tid >> 5;
  const int wm = wid & 3, wn = wid >> 2;          // 4 x 2 wave grid
  const int half = lane >> 4, l16 = lane & 15;
  const int m0 = blockIdx.y * 64, n0 = blockIdx.x * 64;

  v8f acc0 = {0.f,0.f,0.f,0.f,0.f,0.f,0.f,0.f};
  v8f acc1 = acc0;

  for (int k0 = 0; k0 < K; k0 += 32) {
#if HAVE_TDM
    if (wid == 0) {
      // 64x32 bf16 tiles; rows 64B, pad 16B: interval code 3 (16 DW), amount code 3 (4 DW)
      tdm_load_tile_bf16(&A[(size_t)m0 * K + k0],
                         (unsigned)(size_t)(void*)&As[0][0], 32, 64, K, 3, 3);
      tdm_load_tile_bf16(&Bm[(size_t)n0 * K + k0],
                         (unsigned)(size_t)(void*)&Bs[0][0], 32, 64, K, 3, 3);
      __builtin_amdgcn_s_wait_tensorcnt(0);
    }
#else
#pragma unroll
    for (int i = tid; i < 64 * 32; i += 256) {
      int r = i >> 5, c = i & 31;
      As[r][c] = A[(size_t)(m0 + r) * K + (k0 + c)];
      Bs[r][c] = Bm[(size_t)(n0 + r) * K + (k0 + c)];
    }
#endif
    if (k0 + 32 < K) {   // prefetch next K-slice toward L2 (global_prefetch_b8)
      __builtin_prefetch(&A[(size_t)(m0 + (tid & 63)) * K + (k0 + 32)], 0, 1);
      __builtin_prefetch(&Bm[(size_t)(n0 + (tid & 63)) * K + (k0 + 32)], 0, 1);
    }
    __syncthreads();

    BfFrag a;
#pragma unroll
    for (int e = 0; e < 16; ++e) a.u[e] = As[wm * 16 + l16][frag_k(e, half)];
    BfFrag b0, b1;
#pragma unroll
    for (int e = 0; e < 16; ++e) {
      int kk = frag_k(e, half);
      b0.u[e] = Bs[wn * 32 + l16][kk];
      b1.u[e] = Bs[wn * 32 + 16 + l16][kk];
    }
    acc0 = wmma_bf16(a, b0, acc0);
    acc1 = wmma_bf16(a, b1, acc1);
    __syncthreads();
  }

#pragma unroll
  for (int j = 0; j < 8; ++j) {
    int m = m0 + wm * 16 + half * 8 + j;     // C layout: VGPR j -> row j + half*8
    int n = n0 + wn * 32 + l16;
    if (m < M && n < N) {
      if constexpr (OUT_BF16) {
        Cb[(size_t)m * N + n]      = f2bf(acc0[j]);
        Cb[(size_t)m * N + n + 16] = f2bf(acc1[j]);
      } else {
        Cf[(size_t)m * N + n]      = acc0[j];
        Cf[(size_t)m * N + n + 16] = acc1[j];
      }
    }
  }
}

// ---------------------------------------------------------------------------
// RoPE + split qkv[B*S, 3*H] -> q/k/v [B, NH, S, HD] (bf16, fp32 rope math)
// ---------------------------------------------------------------------------
__global__ void rope_split_kernel(const unsigned short* __restrict__ qkv,
                                  const float* __restrict__ cosT,
                                  const float* __restrict__ sinT,
                                  unsigned short* __restrict__ qr,
                                  unsigned short* __restrict__ kr,
                                  unsigned short* __restrict__ vr) {
  int i = blockIdx.x * blockDim.x + threadIdx.x;
  if (i >= BB * SS * NH * HD) return;
  int d = i & (HD - 1);
  int h = (i / HD) % NH;
  int s = (i / (HD * NH)) % SS;
  int b = i / (HD * NH * SS);

  size_t row = (size_t)(b * SS + s) * (3 * HH);
  int col = h * HD + d;
  int pd  = (d < HD / 2) ? d + HD / 2 : d - HD / 2;
  float sgn = (d < HD / 2) ? -1.f : 1.f;

  float c  = cosT[((size_t)b * SS + s) * HD + d];
  float sn = sinT[((size_t)b * SS + s) * HD + d];

  float q  = bf2f(qkv[row + col]);
  float qp = bf2f(qkv[row + h * HD + pd]);
  float k  = bf2f(qkv[row + HH + col]);
  float kp = bf2f(qkv[row + HH + h * HD + pd]);
  float v  = bf2f(qkv[row + 2 * HH + col]);

  size_t o = ((size_t)(b * NH + h) * SS + s) * HD + d;
  qr[o] = f2bf(q * c + sgn * qp * sn);
  kr[o] = f2bf(k * c + sgn * kp * sn);
  vr[o] = f2bf(v);
}

// ---------------------------------------------------------------------------
// Sliding-window attention, flash-style: one block per (b, head, 64-query tile).
// Keys/values: 192-row strip [q0-64, q0+128) covers every query's +/-window band.
// 4 waves; wave w owns query rows [w*16, w*16+16).
// ---------------------------------------------------------------------------
__global__ __launch_bounds__(128)
void swa_attention_kernel(const unsigned short* __restrict__ qr,
                          const unsigned short* __restrict__ kr,
                          const unsigned short* __restrict__ vr,
                          const int* __restrict__ winp,
                          unsigned short* __restrict__ y) {
  __shared__ unsigned short Qs[64][72];
  __shared__ unsigned short Ks[192][72];
  __shared__ unsigned short Vs[192][72];
  __shared__ unsigned short Ps[64][200];

  const int qb = blockIdx.x, h = blockIdx.y, b = blockIdx.z;
  const int tid = threadIdx.x;
  const int lane = tid & 31, wid = tid >> 5;
  const int half = lane >> 4, l16 = lane & 15;
  const int window = winp[0];
  const size_t headBase = (size_t)(b * NH + h) * SS * HD;
  const int q0 = qb * 64;
  const int kstart = q0 - 64;

#if HAVE_TDM
  // Q tile is fully in range: TDM it. Rows 128B, pad 16B into 144B pitch:
  // interval code 4 (32 DW = 128B), amount code 3 (4 DW = 16B).
  if (wid == 0) {
    tdm_load_tile_bf16(&qr[headBase + (size_t)q0 * HD],
                       (unsigned)(size_t)(void*)&Qs[0][0], 64, 64, HD, 4, 3);
  }
#else
  for (int i = tid; i < 64 * HD; i += 128) {
    int r = i >> 6, c = i & 63;
    Qs[r][c] = qr[headBase + (size_t)(q0 + r) * HD + c];
  }
#endif
  // K/V strips need zero-fill for rows outside [0,S): keep these manual.
  for (int i = tid; i < 192 * HD; i += 128) {
    int r = i >> 6, c = i & 63;
    int kg = kstart + r;
    bool ok = (kg >= 0) && (kg < SS);
    Ks[r][c] = ok ? kr[headBase + (size_t)kg * HD + c] : (unsigned short)0;
    Vs[r][c] = ok ? vr[headBase + (size_t)kg * HD + c] : (unsigned short)0;
  }
#if HAVE_TDM
  if (wid == 0) __builtin_amdgcn_s_wait_tensorcnt(0);
#endif
  __syncthreads();

  const int m0 = wid * 16;

  // ---- S = Q K^T : 12 N-tiles of 16 keys, K-dim 64 = 2 WMMA steps ----
  v8f acc[12];
  const v8f vzero = {0.f,0.f,0.f,0.f,0.f,0.f,0.f,0.f};
#pragma unroll
  for (int t = 0; t < 12; ++t) acc[t] = vzero;

#pragma unroll
  for (int ks = 0; ks < 2; ++ks) {
    BfFrag a;
#pragma unroll
    for (int e = 0; e < 16; ++e)
      a.u[e] = Qs[m0 + l16][ks * 32 + frag_k(e, half)];
#pragma unroll
    for (int nt = 0; nt < 12; ++nt) {
      BfFrag bf;
#pragma unroll
      for (int e = 0; e < 16; ++e)
        bf.u[e] = Ks[nt * 16 + l16][ks * 32 + frag_k(e, half)];
      acc[nt] = wmma_bf16(a, bf, acc[nt]);
    }
  }

  // ---- masked softmax across the 192-key strip ----
  const float scale = 0.125f;   // 1/sqrt(64)
  float rmax[8], rsum[8];
#pragma unroll
  for (int j = 0; j < 8; ++j) { rmax[j] = -3.0e38f; rsum[j] = 0.f; }

#pragma unroll
  for (int nt = 0; nt < 12; ++nt) {
    int kg = kstart + nt * 16 + l16;
#pragma unroll
    for (int j = 0; j < 8; ++j) {
      int qg = q0 + m0 + half * 8 + j;
      int diff = qg - kg; if (diff < 0) diff = -diff;
      bool valid = (kg >= 0) && (kg < SS) && (diff <= window);
      float s = valid ? acc[nt][j] * scale : -3.0e38f;
      acc[nt][j] = s;
      rmax[j] = fmaxf(rmax[j], s);
    }
  }
  // row reductions: xor masks 1,2,4,8 stay inside each 16-lane N-group
#pragma unroll
  for (int j = 0; j < 8; ++j)
#pragma unroll
    for (int msk = 1; msk <= 8; msk <<= 1)
      rmax[j] = fmaxf(rmax[j], __shfl_xor(rmax[j], msk, 32));

#pragma unroll
  for (int nt = 0; nt < 12; ++nt)
#pragma unroll
    for (int j = 0; j < 8; ++j) {
      float p = __expf(acc[nt][j] - rmax[j]);   // masked -> exp(-huge) = 0
      acc[nt][j] = p;
      rsum[j] += p;
    }
#pragma unroll
  for (int j = 0; j < 8; ++j)
#pragma unroll
    for (int msk = 1; msk <= 8; msk <<= 1)
      rsum[j] += __shfl_xor(rsum[j], msk, 32);

  // normalized probs -> LDS (C-layout -> A-layout transpose through LDS)
#pragma unroll
  for (int nt = 0; nt < 12; ++nt)
#pragma unroll
    for (int j = 0; j < 8; ++j)
      Ps[m0 + half * 8 + j][nt * 16 + l16] = f2bf(acc[nt][j] / rsum[j]);
  __syncthreads();

  // ---- Y = P V : 4 N-tiles of 16 head-dims, K-dim 192 = 6 WMMA steps ----
  v8f out[4];
#pragma unroll
  for (int t = 0; t < 4; ++t) out[t] = vzero;
#pragma unroll
  for (int ks = 0; ks < 6; ++ks) {
    BfFrag a;
#pragma unroll
    for (int e = 0; e < 16; ++e)
      a.u[e] = Ps[m0 + l16][ks * 32 + frag_k(e, half)];
#pragma unroll
    for (int nt = 0; nt < 4; ++nt) {
      BfFrag bf;
#pragma unroll
      for (int e = 0; e < 16; ++e)
        bf.u[e] = Vs[ks * 32 + frag_k(e, half)][nt * 16 + l16];
      out[nt] = wmma_bf16(a, bf, out[nt]);
    }
  }

  // y layout [B, S, NH*HD] so the output GEMM reads it row-major directly
#pragma unroll
  for (int nt = 0; nt < 4; ++nt)
#pragma unroll
    for (int j = 0; j < 8; ++j) {
      int qg = q0 + m0 + half * 8 + j;
      int d  = nt * 16 + l16;
      y[(size_t)(b * SS + qg) * HH + h * HD + d] = f2bf(out[nt][j]);
    }
}

// ---------------------------------------------------------------------------
extern "C" void kernel_launch(void* const* d_in, const int* in_sizes, int n_in,
                              void* d_out, int out_size, void* d_ws, size_t ws_size,
                              hipStream_t stream) {
  (void)in_sizes; (void)n_in; (void)out_size; (void)ws_size;
  const float* x    = (const float*)d_in[0];
  const float* cosT = (const float*)d_in[1];
  const float* sinT = (const float*)d_in[2];
  const float* Wqkv = (const float*)d_in[3];
  const float* Wo   = (const float*)d_in[4];
  const int*   win  = (const int*)d_in[5];

  const int M = BB * SS;  // 8192 rows

  unsigned short* Xb  = (unsigned short*)d_ws;
  unsigned short* Wqb = Xb  + (size_t)M * HH;
  unsigned short* Wob = Wqb + (size_t)3 * HH * HH;
  unsigned short* QKV = Wob + (size_t)HH * HH;
  unsigned short* QR  = QKV + (size_t)M * 3 * HH;
  unsigned short* KR  = QR  + (size_t)M * HH;
  unsigned short* VR  = KR  + (size_t)M * HH;
  unsigned short* Y   = VR  + (size_t)M * HH;

  int n1 = M * HH;
  f32_to_bf16_kernel<<<(n1 + 255) / 256, 256, 0, stream>>>(x, Xb, n1);
  int n2 = 3 * HH * HH;
  f32_to_bf16_kernel<<<(n2 + 255) / 256, 256, 0, stream>>>(Wqkv, Wqb, n2);
  int n3 = HH * HH;
  f32_to_bf16_kernel<<<(n3 + 255) / 256, 256, 0, stream>>>(Wo, Wob, n3);

  dim3 g1(3 * HH / 64, M / 64);  // 36 x 128 tiles
  gemm_bf16_wmma<true><<<g1, 256, 0, stream>>>(Xb, Wqb, nullptr, QKV, M, 3 * HH, HH);

  rope_split_kernel<<<(M * HH + 255) / 256, 256, 0, stream>>>(QKV, cosT, sinT, QR, KR, VR);

  dim3 ga(SS / 64, NH, BB);      // 32 x 12 x 4 query tiles
  swa_attention_kernel<<<ga, 128, 0, stream>>>(QR, KR, VR, win, Y);

  dim3 g2(HH / 64, M / 64);      // 12 x 128 tiles
  gemm_bf16_wmma<false><<<g2, 256, 0, stream>>>(Y, Wob, (float*)d_out, nullptr, M, HH, HH);
}
